// SelfAttention_32220844655287
// MI455X (gfx1250) — compile-verified, matrix operand order checked by reading, not run on hardware
//
#include <hip/hip_runtime.h>

typedef _Float16 v16h __attribute__((ext_vector_type(16)));
typedef _Float16 v8h  __attribute__((ext_vector_type(8)));
typedef float    v8f  __attribute__((ext_vector_type(8)));

#define L_SEQ   4096
#define D_MODEL 768
#define N_HEADS 12
#define E_HEAD  64
#define SCALE_Q 0.125f            // 64^-0.5
#define LOG2E   1.44269504088896f

#ifndef __has_builtin
#define __has_builtin(x) 0
#endif
#if __has_builtin(__builtin_amdgcn_tensor_load_to_lds) && __has_builtin(__builtin_amdgcn_s_wait_tensorcnt)
#define HAVE_TDM 1
#else
#define HAVE_TDM 0
#endif

// ---- fragment loaders -------------------------------------------------------
// 16x32 f16 WMMA A/B fragment from a row-major f16 matrix (global or LDS).
// A: row0+(lane&15) is the M row.  B (via transposed storage): row0+(lane&15) is the N col.
// K chunks per ISA layout: lanes 0-15 hold K {0..7, 16..23}, lanes 16-31 hold {8..15, 24..31}.
static __device__ __forceinline__ v16h frag_f16(const _Float16* __restrict__ base,
                                                int row0, int stride, int k0) {
  int lane = threadIdx.x & 31;
  const _Float16* p = base + (size_t)(row0 + (lane & 15)) * stride + k0 + ((lane >> 4) << 3);
  v8h lo = *(const v8h*)p;
  v8h hi = *(const v8h*)(p + 16);
  v16h r;
#pragma unroll
  for (int i = 0; i < 8; ++i) { r[i] = lo[i]; r[i + 8] = hi[i]; }
  return r;
}

// Same fragment, but source matrix is f32 in memory (cast to f16 on the fly).
static __device__ __forceinline__ v16h frag_f32(const float* __restrict__ base,
                                                int row0, int stride, int k0) {
  int lane = threadIdx.x & 31;
  const float* p = base + (size_t)(row0 + (lane & 15)) * stride + k0 + ((lane >> 4) << 3);
  v16h r;
#pragma unroll
  for (int i = 0; i < 8; ++i) { r[i] = (_Float16)p[i]; r[i + 8] = (_Float16)p[i + 16]; }
  return r;
}

static __device__ __forceinline__ v8f wmma_f16(v16h a, v16h b, v8f c) {
  return __builtin_amdgcn_wmma_f32_16x16x32_f16(false, a, false, b, (short)0, c, false, false);
}

// ---- Tensor Data Mover: 2D tile load, D# built per ISA 8.3/8.4 --------------
#if HAVE_TDM
typedef unsigned int u32x4 __attribute__((ext_vector_type(4)));
typedef int          i32x8 __attribute__((ext_vector_type(8)));
typedef int          i32x4 __attribute__((ext_vector_type(4)));

// dim0/dim1: tensor extents (elements), tile0/tile1: tile extents, stride0 in elements.
static __device__ __forceinline__ void tdm_load_2d_f16(unsigned lds_addr, const void* gptr,
                                                       unsigned dim0, unsigned dim1,
                                                       unsigned tile0, unsigned tile1,
                                                       unsigned stride0) {
  unsigned long long ga = (unsigned long long)gptr;
  u32x4 g0;
  g0[0] = 1u;                                                   // count=1, user mode
  g0[1] = lds_addr;                                             // LDS byte address
  g0[2] = (unsigned)ga;                                         // global addr [95:64]
  g0[3] = (unsigned)((ga >> 32) & 0x01FFFFFFu) | 0x80000000u;   // addr[120:96] | type=2
  i32x8 g1;
  g1[0] = 0x00010000;                                           // data_size=1 (2 bytes)
  g1[1] = (int)((dim0 & 0xFFFFu) << 16);                        // tensor_dim0[15:0]
  g1[2] = (int)((dim0 >> 16) | ((dim1 & 0xFFFFu) << 16));       // dim0[31:16] | dim1[15:0]
  g1[3] = (int)((dim1 >> 16) | (tile0 << 16));                  // dim1[31:16] | tile_dim0
  g1[4] = (int)tile1;                                           // tile_dim1 (tile_dim2=0)
  g1[5] = (int)stride0;                                         // tensor_dim0_stride[31:0]
  g1[6] = 0;
  g1[7] = 0;
  i32x4 gz = {0, 0, 0, 0};
#if __clang_major__ >= 23
  i32x8 gz8 = {0, 0, 0, 0, 0, 0, 0, 0};
  __builtin_amdgcn_tensor_load_to_lds(g0, g1, gz, gz, gz8, 0);
#else
  __builtin_amdgcn_tensor_load_to_lds(g0, g1, gz, gz, 0);
#endif
}
#endif

// ---- kernel 1: fused QKV projection, f32 -> f16 -----------------------------
// grid = (L/128, 3*H); block = 256 (8 waves).
__global__ __launch_bounds__(256)
void qkv_proj_kernel(const float* __restrict__ X,
                     const float* __restrict__ Wq, const float* __restrict__ bq,
                     const float* __restrict__ Wk, const float* __restrict__ bk,
                     const float* __restrict__ Wv, const float* __restrict__ bv,
                     _Float16* __restrict__ Qh, _Float16* __restrict__ Kh,
                     _Float16* __restrict__ Vt) {
  const int wv   = threadIdx.x >> 5;
  const int lane = threadIdx.x & 31;
  const int m0   = blockIdx.x * 128 + wv * 16;
  const int seg  = blockIdx.y / N_HEADS;     // 0=Q 1=K 2=V
  const int head = blockIdx.y % N_HEADS;
  const float* W = (seg == 0) ? Wq : (seg == 1) ? Wk : Wv;
  const float* B = (seg == 0) ? bq : (seg == 1) ? bk : bv;
  const int col0 = head * E_HEAD;

  v8f acc[4] = {{}, {}, {}, {}};
  for (int k0 = 0; k0 < D_MODEL; k0 += 32) {
    v16h a = frag_f32(X, m0, D_MODEL, k0);
#pragma unroll
    for (int t = 0; t < 4; ++t) {
      v16h b = frag_f32(W, col0 + t * 16, D_MODEL, k0);   // B col = W row (X @ W^T)
      acc[t] = wmma_f16(a, b, acc[t]);
    }
  }

  const int chi = lane >> 4, cl = lane & 15;
  if (seg < 2) {
    _Float16* dst = (seg == 0 ? Qh : Kh) + (size_t)head * L_SEQ * E_HEAD;
#pragma unroll
    for (int t = 0; t < 4; ++t) {
      const int e = t * 16 + cl;
      const float bb = B[col0 + e];
#pragma unroll
      for (int i = 0; i < 8; ++i) {
        const int row = m0 + chi * 8 + i;
        dst[(size_t)row * E_HEAD + e] = (_Float16)(acc[t][i] + bb);
      }
    }
  } else {
    _Float16* dst = Vt + (size_t)head * E_HEAD * L_SEQ;   // transposed: [head][e][l]
#pragma unroll
    for (int t = 0; t < 4; ++t) {
      const int e = t * 16 + cl;
      const float bb = B[col0 + e];
      v8h pk;
#pragma unroll
      for (int i = 0; i < 8; ++i) pk[i] = (_Float16)(acc[t][i] + bb);
      *(v8h*)(dst + (size_t)e * L_SEQ + m0 + chi * 8) = pk;
    }
  }
}

// ---- kernel 2: flash attention, TDM-staged K/V, 32 q-rows per wave ----------
// grid = (L/256, H); block = 256 (8 waves). Key step = 64, double-buffered LDS.
__global__ __launch_bounds__(256)
void flash_attn_kernel(const _Float16* __restrict__ Qh, const _Float16* __restrict__ Kh,
                       const _Float16* __restrict__ Vt, const float* __restrict__ mask,
                       float* __restrict__ out) {
  // [0,8K): Kbuf x2 (64 keys x 64 e, row-major); [8K,16K): Vbuf x2 (64 e x 64 keys);
  // [16K,24K): per-wave 16x64 P tiles, column-major (reused for both q-tiles).
  __shared__ __align__(16) _Float16 smem[2 * 4096 + 2 * 4096 + 8 * 1024];
  const int wv   = threadIdx.x >> 5;
  const int lane = threadIdx.x & 31;
  const int head = blockIdx.y;
  const int q0   = blockIdx.x * 256 + wv * 32;    // 32 query rows per wave
  const _Float16* Q = Qh + (size_t)head * L_SEQ * E_HEAD;
  const _Float16* K = Kh + (size_t)head * L_SEQ * E_HEAD;
  const _Float16* V = Vt + (size_t)head * E_HEAD * L_SEQ;
  _Float16* pb = smem + 16384 + wv * 1024;
  const int chi = lane >> 4, cl = lane & 15;
  const int cb  = chi * 8;

  v16h qa[2][2];
#pragma unroll
  for (int qt = 0; qt < 2; ++qt) {
    qa[qt][0] = frag_f16(Q, q0 + qt * 16, E_HEAD, 0);
    qa[qt][1] = frag_f16(Q, q0 + qt * 16, E_HEAD, 32);
  }

  v8f acc[2][4] = {{{}, {}, {}, {}}, {{}, {}, {}, {}}};
  float mrow[2][8], lrow[2][8];
#pragma unroll
  for (int qt = 0; qt < 2; ++qt)
#pragma unroll
    for (int i = 0; i < 8; ++i) { mrow[qt][i] = -1e30f; lrow[qt][i] = 0.0f; }

  int cur = 0;
#if HAVE_TDM
  if (wv == 0) {
    tdm_load_2d_f16((unsigned)(size_t)(smem), K, 64, L_SEQ, 64, 64, 64);
    tdm_load_2d_f16((unsigned)(size_t)(smem + 8192), V, L_SEQ, 64, 64, 64, L_SEQ);
    __builtin_amdgcn_s_wait_tensorcnt(0);
  }
  __syncthreads();
#endif

  for (int j = 0; j < L_SEQ; j += 64) {
#if HAVE_TDM
    if (wv == 0 && j + 64 < L_SEQ) {   // prefetch next tile into the other buffer
      const int nb = cur ^ 1;
      tdm_load_2d_f16((unsigned)(size_t)(smem + nb * 4096),
                      K + (size_t)(j + 64) * E_HEAD, 64, L_SEQ, 64, 64, 64);
      tdm_load_2d_f16((unsigned)(size_t)(smem + 8192 + nb * 4096),
                      V + (j + 64), L_SEQ, 64, 64, 64, L_SEQ);
    }
#else
    __syncthreads();
    {  // cooperative staging fallback
      _Float16* kd = smem;
      _Float16* vd = smem + 8192;
      const _Float16* gk = K + (size_t)j * E_HEAD;          // contiguous 8KB
      for (int u = threadIdx.x; u < 512; u += 256)
        *(v8h*)(kd + u * 8) = *(const v8h*)(gk + u * 8);
      for (int u = threadIdx.x; u < 512; u += 256) {
        const int e = u >> 3, c = (u & 7) * 8;
        *(v8h*)(vd + e * 64 + c) = *(const v8h*)(V + (size_t)e * L_SEQ + j + c);
      }
      cur = 0;
    }
    __syncthreads();
#endif
    const _Float16* Kt = smem + cur * 4096;
    const _Float16* Vl = smem + 8192 + cur * 4096;

#pragma unroll
    for (int qt = 0; qt < 2; ++qt) {
      // S = Q K^T for 64 keys: 4 column tiles x (K=64 as 2 chunks)
      v8f s[4];
#pragma unroll
      for (int ct = 0; ct < 4; ++ct) {
        v16h kb0 = frag_f16(Kt, ct * 16, 64, 0);
        v16h kb1 = frag_f16(Kt, ct * 16, 64, 32);
        v8f z = {};
        z = wmma_f16(qa[qt][0], kb0, z);
        s[ct] = wmma_f16(qa[qt][1], kb1, z);
      }

      // exp2-domain online softmax over 64 columns
      float p[4][8];
#pragma unroll
      for (int i = 0; i < 8; ++i) {
        const int row = q0 + qt * 16 + chi * 8 + i;
        const float* mp = mask + (size_t)row * L_SEQ + j + cl;
        float v0 = s[0][i] * (SCALE_Q * LOG2E) + mp[0]  * LOG2E;
        float v1 = s[1][i] * (SCALE_Q * LOG2E) + mp[16] * LOG2E;
        float v2 = s[2][i] * (SCALE_Q * LOG2E) + mp[32] * LOG2E;
        float v3 = s[3][i] * (SCALE_Q * LOG2E) + mp[48] * LOG2E;
        float mx = fmaxf(fmaxf(v0, v1), fmaxf(v2, v3));
        mx = fmaxf(mx, __shfl_xor(mx, 1, 16));
        mx = fmaxf(mx, __shfl_xor(mx, 2, 16));
        mx = fmaxf(mx, __shfl_xor(mx, 4, 16));
        mx = fmaxf(mx, __shfl_xor(mx, 8, 16));
        const float mo = mrow[qt][i];
        const float mn = fmaxf(mo, mx);
        const float f  = exp2f(mo - mn);
        const float e0 = exp2f(v0 - mn), e1 = exp2f(v1 - mn);
        const float e2 = exp2f(v2 - mn), e3 = exp2f(v3 - mn);
        float rs = (e0 + e1) + (e2 + e3);
        rs += __shfl_xor(rs, 1, 16);
        rs += __shfl_xor(rs, 2, 16);
        rs += __shfl_xor(rs, 4, 16);
        rs += __shfl_xor(rs, 8, 16);
        mrow[qt][i] = mn;
        lrow[qt][i] = lrow[qt][i] * f + rs;
        acc[qt][0][i] *= f; acc[qt][1][i] *= f; acc[qt][2][i] *= f; acc[qt][3][i] *= f;
        p[0][i] = e0; p[1][i] = e1; p[2][i] = e2; p[3][i] = e3;
      }

      // C-layout -> A-layout via wave-private LDS (in-order DS, no barrier)
#pragma unroll
      for (int ct = 0; ct < 4; ++ct) {
        v8h pk;
#pragma unroll
        for (int i = 0; i < 8; ++i) pk[i] = (_Float16)p[ct][i];
        *(v8h*)(pb + (ct * 16 + cl) * 16 + cb) = pk;
      }
      v16h pa0, pa1;
#pragma unroll
      for (int i = 0; i < 8; ++i) {
        pa0[i]     = pb[(cb + i) * 16 + cl];
        pa0[i + 8] = pb[(cb + 16 + i) * 16 + cl];
        pa1[i]     = pb[(32 + cb + i) * 16 + cl];
        pa1[i + 8] = pb[(32 + cb + 16 + i) * 16 + cl];
      }

      // O += P V
#pragma unroll
      for (int t = 0; t < 4; ++t)
        acc[qt][t] = wmma_f16(pa0, frag_f16(Vl, t * 16, 64, 0),  acc[qt][t]);
#pragma unroll
      for (int t = 0; t < 4; ++t)
        acc[qt][t] = wmma_f16(pa1, frag_f16(Vl, t * 16, 64, 32), acc[qt][t]);
    }

#if HAVE_TDM
    if (wv == 0 && j + 64 < L_SEQ) __builtin_amdgcn_s_wait_tensorcnt(0);
    __syncthreads();
    cur ^= 1;
#endif
  }

  // epilogue: normalize and write fp32 output [l][h*64+e]
#pragma unroll
  for (int qt = 0; qt < 2; ++qt)
#pragma unroll
    for (int t = 0; t < 4; ++t)
#pragma unroll
      for (int i = 0; i < 8; ++i) {
        const int row = q0 + qt * 16 + chi * 8 + i;
        out[(size_t)row * D_MODEL + head * E_HEAD + t * 16 + cl] = acc[qt][t][i] / lrow[qt][i];
      }
}

// ---- launch -----------------------------------------------------------------
extern "C" void kernel_launch(void* const* d_in, const int* in_sizes, int n_in,
                              void* d_out, int out_size, void* d_ws, size_t ws_size,
                              hipStream_t stream) {
  const float* X    = (const float*)d_in[0];
  const float* mask = (const float*)d_in[1];
  const float* Wq   = (const float*)d_in[2];
  const float* bq   = (const float*)d_in[3];
  const float* Wk   = (const float*)d_in[4];
  const float* bk   = (const float*)d_in[5];
  const float* Wv   = (const float*)d_in[6];
  const float* bv   = (const float*)d_in[7];
  float* out        = (float*)d_out;

  const size_t per = (size_t)N_HEADS * L_SEQ * E_HEAD;  // 3,145,728 f16 each
  _Float16* Qh = (_Float16*)d_ws;
  _Float16* Kh = Qh + per;
  _Float16* Vt = Kh + per;

  dim3 gp(L_SEQ / 128, 3 * N_HEADS);
  qkv_proj_kernel<<<gp, 256, 0, stream>>>(X, Wq, bq, Wk, bk, Wv, bv, Qh, Kh, Vt);

  dim3 ga(L_SEQ / 256, N_HEADS);
  flash_attn_kernel<<<ga, 256, 0, stream>>>(Qh, Kh, Vt, mask, out);
}
